// SubstractLayer_73461120631381
// MI455X (gfx1250) — matrix operand order repaired and checked by reading, not compiled, not run on hardware
//
#include <hip/hip_runtime.h>

typedef __attribute__((ext_vector_type(16))) _Float16 v16h;
typedef __attribute__((ext_vector_type(8)))  _Float16 v8h;
typedef __attribute__((ext_vector_type(8)))  float    v8f;
typedef __attribute__((ext_vector_type(2)))  float    v2f;

#define B_    2
#define N_    8192
#define M_    8192
#define D_    64
#define WAVES 8
#define SPLIT 4    // M-dimension split factor (per-wave strip = M/SPLIT)

// ---------------------------------------------------------------------------
// Kernel 1: normalize rows. Feats -> unit f16 rows; coords -> unit f32 rows
// padded to 4 elements (for the 16x16x4 f32 WMMA).
// ---------------------------------------------------------------------------
__global__ __launch_bounds__(256)
void cosarg_prep_kernel(const float* __restrict__ coords_a,
                        const float* __restrict__ coords_b,
                        const float* __restrict__ feats_a,
                        const float* __restrict__ feats_b,
                        _Float16* __restrict__ fa16,
                        _Float16* __restrict__ fb16,
                        float* __restrict__ caf,
                        float* __restrict__ cbf)
{
    int tid = blockIdx.x * blockDim.x + threadIdx.x;
    if (tid >= B_ * N_) return;

    #pragma unroll
    for (int side = 0; side < 2; ++side) {
        const float* src = (side == 0 ? feats_a : feats_b) + (size_t)tid * D_;
        float v[D_];
        float ss = 0.f;
        #pragma unroll
        for (int i = 0; i < D_; ++i) { v[i] = src[i]; ss += v[i] * v[i]; }
        float inv = rsqrtf(ss);
        _Float16* dst = (side == 0 ? fa16 : fb16) + (size_t)tid * D_;
        #pragma unroll
        for (int c = 0; c < D_ / 8; ++c) {
            v8h h;
            #pragma unroll
            for (int i = 0; i < 8; ++i) h[i] = (_Float16)(v[c * 8 + i] * inv);
            *(v8h*)(dst + c * 8) = h;
        }
    }

    #pragma unroll
    for (int side = 0; side < 2; ++side) {
        const float* src = (side == 0 ? coords_a : coords_b) + (size_t)tid * 3;
        float x = src[0], y = src[1], z = src[2];
        float inv = rsqrtf(x * x + y * y + z * z);
        float* dst = (side == 0 ? caf : cbf) + (size_t)tid * 4;
        float4 o = make_float4(x * inv, y * inv, z * inv, 0.0f);
        *(float4*)dst = o;
    }
}

// ---------------------------------------------------------------------------
// WMMA tile step: 3 independent WMMAs (coords f32 K=4; feats f16 K=32 x2)
// + fused clamp/product/argmin update.
// ---------------------------------------------------------------------------
__device__ __forceinline__
void tile_update(v2f caA, v16h faA0, v16h faA1,
                 v2f cbB, v16h fb0, v16h fb1,
                 int m, float best[8], int bestm[8])
{
    v8f zero = {};
    v8f cc = __builtin_amdgcn_wmma_f32_16x16x4_f32(
        false, caA, false, cbB, (short)0, zero, false, false);
    v8f f0 = __builtin_amdgcn_wmma_f32_16x16x32_f16(
        false, faA0, false, fb0, (short)0, zero, false, false);
    v8f f1 = __builtin_amdgcn_wmma_f32_16x16x32_f16(
        false, faA1, false, fb1, (short)0, zero, false, false);

    #pragma unroll
    for (int r = 0; r < 8; ++r) {
        float fcr = f0[r] + f1[r];
        float ccr = cc[r];
        float dc = 1.0f - (ccr > 0.0f ? ccr : 0.0f);
        float df = 1.0f - (fcr > 0.0f ? fcr : 0.0f);
        float d  = dc * df;
        if (d < best[r]) { best[r] = d; bestm[r] = m; }
    }
}

// ---------------------------------------------------------------------------
// Kernel 2: fused cos-dist product + per-strip argmin via WMMA.
// One wave owns (n-tile, M-strip); 4096 waves. Mapping puts all waves of a
// block (and neighboring blocks) on the SAME M-strip so their B-fragment
// streams are identical -> WGP$/L2 broadcast hits.
// ---------------------------------------------------------------------------
__global__ __launch_bounds__(32 * WAVES)
void cosarg_wmma_kernel(const _Float16* __restrict__ fa16,
                        const _Float16* __restrict__ fb16,
                        const float* __restrict__ caf,
                        const float* __restrict__ cbf,
                        float* __restrict__ minv_ws,
                        int*   __restrict__ minm_ws)
{
    const int lane = threadIdx.x & 31;
    const int wave = threadIdx.x >> 5;
    const int t    = blockIdx.x * WAVES + wave;       // 0 .. B*(N/16)*SPLIT-1
    const int nt     = t & 1023;                      // B*(N/16) - 1
    const int split  = t >> 10;                       // same split per block
    const int b      = nt >> 9;                       // N_/16 == 512
    const int tile_n = nt & 511;

    const int col  = lane & 15;
    const int half = lane >> 4;

    // ---- A fragments (invariant over the M strip) ----
    const int a_row = tile_n * 16 + col;
    const int k0 = half * 16;

    const _Float16* fa_row = fa16 + ((size_t)b * N_ + a_row) * D_;
    v8h lo, hi;
    lo = *(const v8h*)(fa_row + k0);
    hi = *(const v8h*)(fa_row + k0 + 8);
    v16h faA0 = __builtin_shufflevector(lo, hi, 0,1,2,3,4,5,6,7,8,9,10,11,12,13,14,15);
    lo = *(const v8h*)(fa_row + 32 + k0);
    hi = *(const v8h*)(fa_row + 32 + k0 + 8);
    v16h faA1 = __builtin_shufflevector(lo, hi, 0,1,2,3,4,5,6,7,8,9,10,11,12,13,14,15);

    // f32 16x4 A fragment: lanes 0-15 hold K0,K1; lanes 16-31 hold K2,K3
    v2f caA = *(const v2f*)(caf + ((size_t)b * N_ + a_row) * 4 + half * 2);

    // ---- running per-slot min / argmin ----
    float best[8];
    int   bestm[8];
    #pragma unroll
    for (int r = 0; r < 8; ++r) { best[r] = 3.0e38f; bestm[r] = 0; }

    const int m_base = split * (M_ / SPLIT);
    const _Float16* fb_ptr = fb16 + ((size_t)b * M_ + m_base + col) * D_ + half * 16;
    const float*    cb_ptr = cbf  + ((size_t)b * M_ + m_base + col) * 4 + half * 2;

    const int MT = M_ / (16 * SPLIT);                 // 128 tiles per strip

    // prime the pipeline
    v16h fb0c = *(const v16h*)(fb_ptr);
    v16h fb1c = *(const v16h*)(fb_ptr + 32);
    v2f  cbc  = *(const v2f*)(cb_ptr);

    int m = m_base + col;
    #pragma unroll 2
    for (int mt = 0; mt < MT - 1; ++mt) {
        fb_ptr += 16 * D_;
        cb_ptr += 16 * 4;
        // prefetch next tile while computing current
        v16h fb0n = *(const v16h*)(fb_ptr);
        v16h fb1n = *(const v16h*)(fb_ptr + 32);
        v2f  cbn  = *(const v2f*)(cb_ptr);

        tile_update(caA, faA0, faA1, cbc, fb0c, fb1c, m, best, bestm);
        m += 16;

        fb0c = fb0n; fb1c = fb1n; cbc = cbn;
    }
    tile_update(caA, faA0, faA1, cbc, fb0c, fb1c, m, best, bestm);

    // ---- butterfly argmin over the 16 columns inside each half-wave ----
    #pragma unroll
    for (int mask = 1; mask < 16; mask <<= 1) {
        #pragma unroll
        for (int r = 0; r < 8; ++r) {
            float ov = __shfl_xor(best[r], mask, 32);
            int   om = __shfl_xor(bestm[r], mask, 32);
            if (ov < best[r] || (ov == best[r] && om < bestm[r])) {
                best[r] = ov; bestm[r] = om;
            }
        }
    }
    // lane 0 holds rows 0..7 of the tile, lane 16 holds rows 8..15

    if (col == 0) {
        size_t base = ((size_t)(b * N_ + tile_n * 16 + half * 8)) * SPLIT + split;
        #pragma unroll
        for (int r = 0; r < 8; ++r) {
            minv_ws[base + (size_t)r * SPLIT] = best[r];
            minm_ws[base + (size_t)r * SPLIT] = bestm[r];
        }
    }
}

// ---------------------------------------------------------------------------
// Kernel 3: merge the SPLIT candidates per row, gather feats_b, concat, write.
// ---------------------------------------------------------------------------
__global__ __launch_bounds__(32 * WAVES)
void cosarg_finalize_kernel(const float* __restrict__ feats_a,
                            const float* __restrict__ feats_b,
                            const float* __restrict__ minv_ws,
                            const int*   __restrict__ minm_ws,
                            float* __restrict__ out)
{
    const int row  = blockIdx.x * WAVES + (threadIdx.x >> 5);  // 0 .. B*N-1
    const int lane = threadIdx.x & 31;

    const float* pv = minv_ws + (size_t)row * SPLIT;
    const int*   pm = minm_ws + (size_t)row * SPLIT;
    float bv = pv[0];
    int   bm = pm[0];
    #pragma unroll
    for (int s = 1; s < SPLIT; ++s) {
        float v = pv[s];
        int   m = pm[s];
        if (v < bv || (v == bv && m < bm)) { bv = v; bm = m; }
    }

    const int b   = row >> 13;        // N_ == 8192
    const int col = lane & 15;
    float* orow = out + (size_t)row * (2 * D_);
    if (lane < 16) {
        float4 v = *(const float4*)(feats_a + (size_t)row * D_ + col * 4);
        *(float4*)(orow + col * 4) = v;
    } else {
        float4 v = *(const float4*)(feats_b + ((size_t)b * M_ + bm) * D_ + col * 4);
        *(float4*)(orow + D_ + col * 4) = v;
    }
}

// ---------------------------------------------------------------------------
extern "C" void kernel_launch(void* const* d_in, const int* in_sizes, int n_in,
                              void* d_out, int out_size, void* d_ws, size_t ws_size,
                              hipStream_t stream) {
    const float* coords_a = (const float*)d_in[0];  // [B,N,3]
    const float* coords_b = (const float*)d_in[1];  // [B,M,3]
    const float* feats_a  = (const float*)d_in[2];  // [B,N,64]
    const float* feats_b  = (const float*)d_in[3];  // [B,M,64]
    float* out = (float*)d_out;                     // [B,N,128]

    // workspace: f16 feats (4 MB) + f32 padded coords (512 KB) + results (512 KB)
    _Float16* fa16 = (_Float16*)d_ws;
    _Float16* fb16 = fa16 + (size_t)B_ * N_ * D_;
    float* caf = (float*)(fb16 + (size_t)B_ * M_ * D_);
    float* cbf = caf + (size_t)B_ * N_ * 4;
    float* minv_ws = cbf + (size_t)B_ * M_ * 4;
    int*   minm_ws = (int*)(minv_ws + (size_t)B_ * N_ * SPLIT);

    // Kernel 1: normalization / conversion
    {
        int threads = B_ * N_;
        dim3 block(256), grid((threads + 255) / 256);
        cosarg_prep_kernel<<<grid, block, 0, stream>>>(
            coords_a, coords_b, feats_a, feats_b, fa16, fb16, caf, cbf);
    }

    // Kernel 2: fused WMMA distance + per-strip argmin
    {
        int n_waves = B_ * (N_ / 16) * SPLIT;               // 4096
        dim3 block(32 * WAVES), grid(n_waves / WAVES);      // 512 blocks
        cosarg_wmma_kernel<<<grid, block, 0, stream>>>(
            fa16, fb16, caf, cbf, minv_ws, minm_ws);
    }

    // Kernel 3: merge splits + gather/concat
    {
        int rows = B_ * N_;                                 // 16384
        dim3 block(32 * WAVES), grid(rows / WAVES);         // 2048 blocks
        cosarg_finalize_kernel<<<grid, block, 0, stream>>>(
            feats_a, feats_b, minv_ws, minm_ws, out);
    }
}
